// PartialConvLayer_56040733278871
// MI455X (gfx1250) — matrix operand order repaired and checked by reading, not compiled, not run on hardware
//
#include <hip/hip_runtime.h>

typedef __attribute__((ext_vector_type(16))) __bf16 v16bf;
typedef __attribute__((ext_vector_type(8)))  __bf16 v8bf;
typedef __attribute__((ext_vector_type(8)))  float  v8f;
typedef __attribute__((ext_vector_type(4)))  float  v4f;

#define NB    8
#define CIN   64
#define COUT  128
#define HDIM  256
#define WDIM  256
#define HW    (HDIM*WDIM)          // 65536
#define KTOT  576                  // 9 taps * 64 cin
#define CPAD  72                   // padded cin stride in LDS (16B-aligned rows, bank spread)
#define NSEG  64

// ---------------- K1: xm (channels-last bf16) = x*mask, msum = sum_cin(mask) ----
// 8 threads per pixel; each handles an 8-cin chunk -> one coalesced b128 store.
__global__ __launch_bounds__(256) void k_prep(const float* __restrict__ x,
                                              const float* __restrict__ mask,
                                              __bf16* __restrict__ xmcl,
                                              float* __restrict__ msum) {
  int gid = blockIdx.x * 256 + threadIdx.x;
  int chunk = gid & 7;            // cin chunk of 8
  int pid   = gid >> 3;           // n*HW + s
  int n = pid >> 16;
  int s = pid & (HW - 1);
  float acc = 0.f;
  v8bf pk;
  #pragma unroll
  for (int j = 0; j < 8; ++j) {
    int c = chunk * 8 + j;
    size_t idx = ((size_t)(n * CIN + c) << 16) + s;
    float m = mask[idx];
    float v = x[idx];
    pk[j] = (__bf16)(v * m);
    acc += m;
  }
  *(v8bf*)(xmcl + ((size_t)pid << 6) + chunk * 8) = pk;
  // reduce the 8 chunk-partials of this pixel (lanes 0..7 of each group)
  acc += __shfl_xor(acc, 1);
  acc += __shfl_xor(acc, 2);
  acc += __shfl_xor(acc, 4);
  if (chunk == 0) msum[pid] = acc;
}

// ---------------- K1w: W (Cout,Cin,3,3) -> bf16 [cout][tap][cin] ----------------
__global__ __launch_bounds__(256) void k_wprep(const float* __restrict__ Wsrc,
                                               __bf16* __restrict__ Wr) {
  int t = blockIdx.x * 256 + threadIdx.x;
  if (t >= COUT * KTOT) return;
  int cout = t / KTOT;
  int r    = t - cout * KTOT;
  int tap  = r >> 6;
  int cin  = r & 63;
  Wr[t] = (__bf16)Wsrc[(cout * CIN + cin) * 9 + tap];
}

// ---------------- K1u: update = 3x3 box-sum of msum (zero padded) ----------------
__global__ __launch_bounds__(256) void k_update(const float* __restrict__ msum,
                                                float* __restrict__ us) {
  int pid = blockIdx.x * 256 + threadIdx.x;
  int n = pid >> 16;
  int s = pid & (HW - 1);
  int h = s >> 8, w = s & 255;
  float acc = 0.f;
  for (int dh = -1; dh <= 1; ++dh)
    for (int dw = -1; dw <= 1; ++dw) {
      int hh = h + dh, ww = w + dw;
      if (hh >= 0 && hh < HDIM && ww >= 0 && ww < WDIM)
        acc += msum[(n << 16) + (hh << 8) + ww];
    }
  us[pid] = acc;
}

// ---------------- K2: implicit-GEMM conv via bf16 WMMA + async LDS staging -------
// GEMM: D(M=128 cout, N=128 pixels) = A(Wr, MxK) * B(patches, KxN), K = 576.
// Block: 1 image row x 128 pixels, all 128 couts. 8 waves = 2(M) x 4(N).
__global__ __launch_bounds__(256) void k_conv_wmma(
    const __bf16* __restrict__ xmcl, const __bf16* __restrict__ Wr,
    const float* __restrict__ us, const float* __restrict__ bias,
    float* __restrict__ outPre) {
  __shared__ __attribute__((aligned(16))) __bf16 Ls[3 * 130 * CPAD];
  __shared__ float sBias[COUT];

  int tid = threadIdx.x;
  int n = blockIdx.z, h = blockIdx.y, w0 = blockIdx.x * 128;

  // Stage haloed input block into LDS [dh][wh][cin] (CPAD stride, 16B chunks).
  // In-bounds chunks: hardware async copy global->LDS. OOB halo: zero-fill.
  for (int i = tid; i < 3 * 130 * 8; i += 256) {
    int chunk = i & 7;            // 8-cin chunk
    int t = i >> 3;               // 0..389
    int dh = t / 130;
    int wh = t - dh * 130;
    int hh = h + dh - 1;
    int ww = w0 + wh - 1;
    unsigned lds_off =
        (unsigned)(size_t)(&Ls[(dh * 130 + wh) * CPAD + chunk * 8]);
    if (hh >= 0 && hh < HDIM && ww >= 0 && ww < WDIM) {
      unsigned long long gaddr = (unsigned long long)(size_t)(
          xmcl + ((((size_t)n << 16) + (hh << 8) + ww) << 6) + chunk * 8);
      asm volatile("global_load_async_to_lds_b128 %0, %1, off"
                   :: "v"(lds_off), "v"(gaddr) : "memory");
    } else {
      v8bf z = {};
      *(v8bf*)(&Ls[(dh * 130 + wh) * CPAD + chunk * 8]) = z;
    }
  }
  if (tid < COUT) sBias[tid] = bias[tid];
  asm volatile("s_wait_asynccnt 0x0" ::: "memory");
  __syncthreads();

  int wid = tid >> 5, lane = tid & 31;
  int wave_m = wid >> 2;   // 0..1 : 64 couts each
  int wave_n = wid & 3;    // 0..3 : 32 pixels each
  int lhi = lane >> 4, llo = lane & 15;

  v8f acc[4][2] = {};

  #pragma unroll
  for (int tap = 0; tap < 9; ++tap) {
    int dh = tap / 3;
    int dw = tap - dh * 3;
    #pragma unroll
    for (int kc = 0; kc < 2; ++kc) {
      int k0 = kc * 32;
      // B fragments (patches): lane holds col N = pixel, 16 contiguous K values
      v16bf bfr[2];
      #pragma unroll
      for (int nf = 0; nf < 2; ++nf) {
        int p = wave_n * 32 + nf * 16 + llo;
        const __bf16* src = &Ls[(dh * 130 + p + dw) * CPAD + k0 + lhi * 16];
        v8bf lo = *(const v8bf*)src;
        v8bf hi = *(const v8bf*)(src + 8);
        bfr[nf] = __builtin_shufflevector(lo, hi, 0,1,2,3,4,5,6,7,8,9,10,11,12,13,14,15);
      }
      // A fragments (weights): lane holds row M = cout, 16 contiguous K values
      v16bf afr[4];
      #pragma unroll
      for (int mf = 0; mf < 4; ++mf) {
        int c = wave_m * 64 + mf * 16 + llo;
        const __bf16* src = &Wr[(c * 9 + tap) * 64 + k0 + lhi * 16];
        v8bf lo = *(const v8bf*)src;
        v8bf hi = *(const v8bf*)(src + 8);
        afr[mf] = __builtin_shufflevector(lo, hi, 0,1,2,3,4,5,6,7,8,9,10,11,12,13,14,15);
      }
      #pragma unroll
      for (int mf = 0; mf < 4; ++mf)
        #pragma unroll
        for (int nf = 0; nf < 2; ++nf)
          acc[mf][nf] = __builtin_amdgcn_wmma_f32_16x16x32_bf16(
              false, afr[mf], false, bfr[nf], (short)0, acc[mf][nf], false, false);
    }
  }

  // Epilogue: out_pre = (conv * mask_ratio + b) * update_clipped
  #pragma unroll
  for (int nf = 0; nf < 2; ++nf) {
    int p = wave_n * 32 + nf * 16 + llo;
    int pix = (n << 16) + (h << 8) + w0 + p;
    float u  = us[pix];
    float uc = fminf(fmaxf(u, 0.f), 1.f);
    float mr = (576.0f / (u + 1e-6f)) * uc;
    #pragma unroll
    for (int mf = 0; mf < 4; ++mf) {
      int cb = wave_m * 64 + mf * 16 + lhi * 8;
      #pragma unroll
      for (int r = 0; r < 8; ++r) {
        int c = cb + r;
        float v = (acc[mf][nf][r] * mr + sBias[c]) * uc;
        outPre[((size_t)(n * COUT + c) << 16) + (h << 8) + w0 + p] = v;
      }
    }
  }
}

// ---------------- K3: per-channel partial sums (deterministic) ----------------
__global__ __launch_bounds__(256) void k_stats(const float* __restrict__ outPre,
                                               float* __restrict__ part) {
  __shared__ float sS[256], sQ[256];
  int c = blockIdx.x;
  int seg = blockIdx.y;
  int nimg = seg >> 3;
  int s0 = (seg & 7) * 8192;
  const v4f* basep = (const v4f*)(outPre + ((size_t)(nimg * COUT + c) << 16) + s0);
  float S = 0.f, Q = 0.f;
  for (int j = threadIdx.x; j < 2048; j += 256) {
    v4f v = basep[j];
    #pragma unroll
    for (int k = 0; k < 4; ++k) { S += v[k]; Q += v[k] * v[k]; }
  }
  sS[threadIdx.x] = S; sQ[threadIdx.x] = Q;
  __syncthreads();
  for (int st = 128; st > 0; st >>= 1) {
    if (threadIdx.x < st) {
      sS[threadIdx.x] += sS[threadIdx.x + st];
      sQ[threadIdx.x] += sQ[threadIdx.x + st];
    }
    __syncthreads();
  }
  if (threadIdx.x == 0) {
    part[(c * NSEG + seg) * 2 + 0] = sS[0];
    part[(c * NSEG + seg) * 2 + 1] = sQ[0];
  }
}

// ---------------- K4: fold BN into per-channel scale/shift ----------------
__global__ void k_finalize(const float* __restrict__ part,
                           const float* __restrict__ gamma,
                           const float* __restrict__ beta,
                           float* __restrict__ ss) {
  int c = threadIdx.x;
  if (c >= COUT) return;
  float S = 0.f, Q = 0.f;
  for (int seg = 0; seg < NSEG; ++seg) {
    S += part[(c * NSEG + seg) * 2 + 0];
    Q += part[(c * NSEG + seg) * 2 + 1];
  }
  const float cnt = (float)((size_t)NB * HW);
  float mean = S / cnt;
  float var  = Q / cnt - mean * mean;
  float rstd = rsqrtf(var + 1e-5f);
  float sc = gamma[c] * rstd;
  ss[c * 2 + 0] = sc;
  ss[c * 2 + 1] = beta[c] - mean * sc;
}

// ---------------- K5: in-place BN+ReLU, write update_full ----------------
__global__ __launch_bounds__(256) void k_apply(float* __restrict__ out,
                                               const float* __restrict__ ss,
                                               const float* __restrict__ us,
                                               float* __restrict__ updFull) {
  size_t i4 = (size_t)blockIdx.x * 256 + threadIdx.x;
  size_t base = i4 * 4;
  int c    = (int)((base >> 16) & (COUT - 1));
  int nimg = (int)(base >> 23);
  int s    = (int)(base & (HW - 1));
  float sc = ss[c * 2 + 0], sh = ss[c * 2 + 1];
  v4f v = *(v4f*)(out + base);
  v4f u4 = *(const v4f*)(us + ((size_t)nimg << 16) + s);
  v4f uv;
  #pragma unroll
  for (int j = 0; j < 4; ++j) {
    float t = v[j] * sc + sh;
    v[j] = t > 0.f ? t : 0.f;
    uv[j] = fminf(fmaxf(u4[j], 0.f), 1.f);
  }
  *(v4f*)(out + base) = v;
  *(v4f*)(updFull + base) = uv;
}

extern "C" void kernel_launch(void* const* d_in, const int* in_sizes, int n_in,
                              void* d_out, int out_size, void* d_ws, size_t ws_size,
                              hipStream_t stream) {
  const float* x     = (const float*)d_in[0];
  const float* mask  = (const float*)d_in[1];
  const float* Wsrc  = (const float*)d_in[2];
  const float* bias  = (const float*)d_in[3];
  const float* gamma = (const float*)d_in[4];
  const float* beta  = (const float*)d_in[5];

  // workspace layout (bytes)
  constexpr size_t OFF_XM   = 0;                                    // 67,108,864
  constexpr size_t OFF_WR   = OFF_XM + (size_t)NB * CIN * HW * 2;   // +147,456
  constexpr size_t OFF_MS   = OFF_WR + (size_t)COUT * KTOT * 2;     // +2,097,152
  constexpr size_t OFF_US   = OFF_MS + (size_t)NB * HW * 4;         // +2,097,152
  constexpr size_t OFF_PART = OFF_US + (size_t)NB * HW * 4;         // +65,536
  constexpr size_t OFF_SS   = OFF_PART + (size_t)COUT * NSEG * 2 * 4;

  char* ws = (char*)d_ws;
  __bf16* xmcl = (__bf16*)(ws + OFF_XM);
  __bf16* Wr   = (__bf16*)(ws + OFF_WR);
  float*  msum = (float*)(ws + OFF_MS);
  float*  us   = (float*)(ws + OFF_US);
  float*  part = (float*)(ws + OFF_PART);
  float*  ss   = (float*)(ws + OFF_SS);

  float* outF = (float*)d_out;
  float* updF = outF + (size_t)NB * COUT * HW;

  k_prep<<<(NB * HW * 8) / 256, 256, 0, stream>>>(x, mask, xmcl, msum);
  k_wprep<<<(COUT * KTOT + 255) / 256, 256, 0, stream>>>(Wsrc, Wr);
  k_update<<<(NB * HW) / 256, 256, 0, stream>>>(msum, us);

  dim3 gconv(WDIM / 128, HDIM, NB);
  k_conv_wmma<<<gconv, 256, 0, stream>>>(xmcl, Wr, us, bias, outF);

  dim3 gstat(COUT, NSEG);
  k_stats<<<gstat, 256, 0, stream>>>(outF, part);
  k_finalize<<<1, COUT, 0, stream>>>(part, gamma, beta, ss);

  size_t total = (size_t)NB * COUT * HW;
  k_apply<<<(unsigned)(total / 4 / 256), 256, 0, stream>>>(outF, ss, us, updF);
}
